// GeneralGNN_44487271252562
// MI455X (gfx1250) — matrix operand order repaired and checked by reading, not compiled, not run on hardware
//
#include <hip/hip_runtime.h>
#include <cstdint>
#include <cstddef>

#define NN 100000     // nodes
#define NE 1600000    // edges
#define HID 128

typedef __bf16 bf16_t;
typedef __attribute__((ext_vector_type(8)))  __bf16 v8bf;
typedef __attribute__((ext_vector_type(16))) __bf16 v16bf;
typedef __attribute__((ext_vector_type(8)))  float  v8f;

// ---------------------------------------------------------------------------
// WMMA fragment loaders (bf16, 16x16x32, wave32)
//   A (16x32, MxK):  lane<16 holds K = kb..kb+7 and kb+16..kb+23 (kb=0);
//                    lane>=16 same rows, kb=8.   -> two 16B loads, concat.
//   B (32x16, KxN):  same K mapping as A (kb=0 / kb=8, K = kb..kb+7, kb+16..kb+23),
//                    column = lane&15 (W stored transposed [N,K]).
// ---------------------------------------------------------------------------
static __device__ __forceinline__ v16bf ld_a_frag(const bf16_t* p) {
  v8bf lo = *(const v8bf*)(p);
  v8bf hi = *(const v8bf*)(p + 16);
  return __builtin_shufflevector(lo, hi, 0,1,2,3,4,5,6,7,8,9,10,11,12,13,14,15);
}
static __device__ __forceinline__ v16bf ld_b_frag(const bf16_t* p) {
  v8bf lo = *(const v8bf*)(p);
  v8bf hi = *(const v8bf*)(p + 16);
  return __builtin_shufflevector(lo, hi, 0,1,2,3,4,5,6,7,8,9,10,11,12,13,14,15);
}

// ---------------------------------------------------------------------------
// Tiled bf16 WMMA GEMM:  C[M,N] = act( A[M,K] @ Wt[N,K]^T + bias )
// Block = 256 threads (8 waves), block tile 128(M) x 128(N), wave tile 32x64.
// Optional f32 output (Cf, ldc) and/or bf16 output (Cb, ldcb).
// act: 0 = none, 1 = ReLU, 2 = LeakyReLU(0.1)
// ---------------------------------------------------------------------------
__global__ __launch_bounds__(256)
void k_gemm_bf16_wmma(const bf16_t* __restrict__ A, int lda,
                      const bf16_t* __restrict__ Wt,          // [N,K] row-major
                      const float*  __restrict__ bias,        // [N] or null
                      float*  __restrict__ Cf, int ldc,       // optional f32 out
                      bf16_t* __restrict__ Cb, int ldcb,      // optional bf16 out
                      int M, int K, int act)
{
  const int lane = threadIdx.x & 31;
  const int wave = threadIdx.x >> 5;
  const int wm   = wave & 3;          // 4 waves along M
  const int wn   = wave >> 2;         // 2 waves along N
  const int half = lane >> 4;
  const int r    = lane & 15;

  const int mbase = blockIdx.x * 128 + wm * 32;
  const int nbase = blockIdx.y * 128 + wn * 64;

  v8f acc[2][4];
#pragma unroll
  for (int i = 0; i < 2; ++i)
#pragma unroll
    for (int j = 0; j < 4; ++j)
#pragma unroll
      for (int t = 0; t < 8; ++t) acc[i][j][t] = 0.0f;

  int row0 = mbase + r;          if (row0 >= M) row0 = M - 1;
  int row1 = mbase + 16 + r;     if (row1 >= M) row1 = M - 1;
  const bf16_t* aptr0 = A + (size_t)row0 * lda;
  const bf16_t* aptr1 = A + (size_t)row1 * lda;
  const bf16_t* bptr[4];
#pragma unroll
  for (int j = 0; j < 4; ++j) bptr[j] = Wt + (size_t)(nbase + j * 16 + r) * K;

  const int akoff = half * 8;    // A per-lane K offset within 32-step
  const int bkoff = half * 8;    // B per-lane K offset within 32-step

  for (int k0 = 0; k0 < K; k0 += 32) {
    if (k0 + 32 < K) {           // gfx1250 global_prefetch_b8 for next A block
      __builtin_prefetch((const void*)(aptr0 + k0 + 32 + akoff), 0, 3);
      __builtin_prefetch((const void*)(aptr1 + k0 + 32 + akoff), 0, 3);
    }
    v16bf a[2], b[4];
    a[0] = ld_a_frag(aptr0 + k0 + akoff);
    a[1] = ld_a_frag(aptr1 + k0 + akoff);
#pragma unroll
    for (int j = 0; j < 4; ++j) b[j] = ld_b_frag(bptr[j] + k0 + bkoff);
#pragma unroll
    for (int i = 0; i < 2; ++i)
#pragma unroll
      for (int j = 0; j < 4; ++j)
        acc[i][j] = __builtin_amdgcn_wmma_f32_16x16x32_bf16(
            false, a[i], false, b[j], (short)0, acc[i][j], false, false);
  }

  // Epilogue: D layout -> vgpr t, lanes 0-15: M=t, N=lane; lanes 16-31: M=8+t.
#pragma unroll
  for (int i = 0; i < 2; ++i) {
    const int rbase = mbase + i * 16 + half * 8;
#pragma unroll
    for (int j = 0; j < 4; ++j) {
      const int col = nbase + j * 16 + r;
      const float bv = bias ? bias[col] : 0.0f;
#pragma unroll
      for (int t = 0; t < 8; ++t) {
        const int row = rbase + t;
        if (row < M) {
          float v = acc[i][j][t] + bv;
          if (act == 1)      v = v > 0.0f ? v : 0.0f;
          else if (act == 2) v = v > 0.0f ? v : 0.1f * v;
          if (Cf) Cf[(size_t)row * ldc  + col] = v;
          if (Cb) Cb[(size_t)row * ldcb + col] = (bf16_t)v;
        }
      }
    }
  }
}

// ---------------------------------------------------------------------------
// Elementwise / graph kernels
// ---------------------------------------------------------------------------
__global__ void k_f32_to_bf16(const float* __restrict__ in, bf16_t* __restrict__ out, int n) {
  int i = blockIdx.x * blockDim.x + threadIdx.x;
  if (i < n) out[i] = (bf16_t)in[i];
}

// in: [K,N] f32 row-major  ->  out: [N,K] bf16 row-major (transposed weights)
__global__ void k_transpose_bf16(const float* __restrict__ in, bf16_t* __restrict__ out,
                                 int K, int N) {
  int i = blockIdx.x * blockDim.x + threadIdx.x;
  if (i < K * N) {
    int k = i / N, n = i - k * N;
    out[(size_t)n * K + k] = (bf16_t)in[i];
  }
}

__global__ void k_fill_f32(float* __restrict__ p, float v, int n) {
  int i = blockIdx.x * blockDim.x + threadIdx.x;
  if (i < n) p[i] = v;
}

__global__ void k_degree(const int* __restrict__ ei, float* __restrict__ deg, int E) {
  int e = blockIdx.x * blockDim.x + threadIdx.x;
  if (e < E) {
    int d = ei[E + e];  // dst row of edge_index
    __hip_atomic_fetch_add(&deg[d], 1.0f, __ATOMIC_RELAXED, __HIP_MEMORY_SCOPE_AGENT);
  }
}

__global__ void k_dinv(const float* __restrict__ deg, float* __restrict__ dinv, int n) {
  int i = blockIdx.x * blockDim.x + threadIdx.x;
  if (i < n) { float d = deg[i]; dinv[i] = d > 0.0f ? rsqrtf(d) : 0.0f; }
}

// agg[i,:] = dinv[i]^2 * hlin[i,:]   (self-loop contribution, no atomics)
__global__ void k_agg_init(const float* __restrict__ hlin, const float* __restrict__ dinv,
                           float* __restrict__ agg, int n) {
  int i = blockIdx.x * blockDim.x + threadIdx.x;
  if (i < n) {
    float di = dinv[i >> 7];
    agg[i] = di * di * hlin[i];
  }
}

// One wave per edge: float4 gather of h[src], scale by dinv[s]*dinv[d],
// scatter-add into agg[dst] with f32 global atomics.
__global__ __launch_bounds__(256)
void k_scatter(const int* __restrict__ ei, const float* __restrict__ hlin,
               const float* __restrict__ dinv, float* __restrict__ agg, int E) {
  int gw   = (blockIdx.x * blockDim.x + threadIdx.x) >> 5;  // edge id
  int lane = threadIdx.x & 31;
  if (gw >= E) return;
  int s = ei[gw];
  int d = ei[E + gw];
  float nrm = dinv[s] * dinv[d];
  const float4 v = *(const float4*)(hlin + (size_t)s * HID + lane * 4);
  float* base = agg + (size_t)d * HID + lane * 4;
  __hip_atomic_fetch_add(base + 0, nrm * v.x, __ATOMIC_RELAXED, __HIP_MEMORY_SCOPE_AGENT);
  __hip_atomic_fetch_add(base + 1, nrm * v.y, __ATOMIC_RELAXED, __HIP_MEMORY_SCOPE_AGENT);
  __hip_atomic_fetch_add(base + 2, nrm * v.z, __ATOMIC_RELAXED, __HIP_MEMORY_SCOPE_AGENT);
  __hip_atomic_fetch_add(base + 3, nrm * v.w, __ATOMIC_RELAXED, __HIP_MEMORY_SCOPE_AGENT);
}

// embh[i, colofs + c] = bf16(relu(agg[i,c] + bias[c]))
__global__ void k_finalize(const float* __restrict__ agg, const float* __restrict__ bias,
                           bf16_t* __restrict__ embh, int colofs, int n) {
  int i = blockIdx.x * blockDim.x + threadIdx.x;
  if (i < n) {
    int node = i >> 7, c = i & (HID - 1);
    float v = agg[i] + bias[c];
    v = v > 0.0f ? v : 0.0f;
    embh[(size_t)node * 640 + colofs + c] = (bf16_t)v;
  }
}

// ---------------------------------------------------------------------------
// Launcher
// ---------------------------------------------------------------------------
extern "C" void kernel_launch(void* const* d_in, const int* in_sizes, int n_in,
                              void* d_out, int out_size, void* d_ws, size_t ws_size,
                              hipStream_t stream) {
  (void)in_sizes; (void)n_in; (void)out_size; (void)ws_size;

  const float* x      = (const float*)d_in[0];
  const int*   ei     = (const int*)  d_in[1];
  const float* pre_w  = (const float*)d_in[2];
  const float* pre_b  = (const float*)d_in[3];
  const float* conv_w[4] = {(const float*)d_in[4], (const float*)d_in[6],
                            (const float*)d_in[8], (const float*)d_in[10]};
  const float* conv_b[4] = {(const float*)d_in[5], (const float*)d_in[7],
                            (const float*)d_in[9], (const float*)d_in[11]};
  const float* post_w1 = (const float*)d_in[12]; const float* post_b1 = (const float*)d_in[13];
  const float* post_w2 = (const float*)d_in[14]; const float* post_b2 = (const float*)d_in[15];
  const float* post_w3 = (const float*)d_in[16]; const float* post_b3 = (const float*)d_in[17];
  const float* post_w4 = (const float*)d_in[18]; const float* post_b4 = (const float*)d_in[19];

  char* ws = (char*)d_ws;
  size_t off = 0;
  auto carve = [&](size_t bytes) -> void* {
    void* p = ws + off;
    off = (off + bytes + 255) & ~(size_t)255;
    return p;
  };

  float*  deg  = (float*) carve((size_t)NN * 4);
  float*  dinv = (float*) carve((size_t)NN * 4);
  bf16_t* embh = (bf16_t*)carve((size_t)NN * 640 * 2);   // concat features, bf16
  bf16_t* xh   = (bf16_t*)carve((size_t)NN * HID * 2);   // bf16 copy of x
  float*  hlin = (float*) carve((size_t)NN * HID * 4);   // conv linear out, f32
  float*  agg  = (float*) carve((size_t)NN * HID * 4);   // scatter accumulator
  bf16_t* wt_pre = (bf16_t*)carve(128 * 128 * 2);
  bf16_t* wt_conv[4];
  for (int l = 0; l < 4; ++l) wt_conv[l] = (bf16_t*)carve((size_t)128 * 128 * (l + 1) * 2);
  bf16_t* wt_p1 = (bf16_t*)carve(640 * 128 * 2);
  bf16_t* wt_p2 = (bf16_t*)carve(128 * 128 * 2);
  bf16_t* wt_p3 = (bf16_t*)carve(128 * 256 * 2);
  bf16_t* wt_p4 = (bf16_t*)carve(256 * 128 * 2);
  // post-MP temporaries alias dead conv-phase buffers
  bf16_t* t1h = xh;              // [NN,128] bf16 (x consumed after pre GEMM)
  bf16_t* t2h = (bf16_t*)hlin;   // [NN,128] bf16
  bf16_t* t3h = (bf16_t*)agg;    // [NN,256] bf16

  const int T = 256;
  const int nElem = NN * HID;                       // 12.8M
  const int eb = (nElem + T - 1) / T;

  // --- precompute: bf16 conversions / weight transposes ---
  k_f32_to_bf16<<<eb, T, 0, stream>>>(x, xh, nElem);
  k_transpose_bf16<<<(128 * 128 + T - 1) / T, T, 0, stream>>>(pre_w, wt_pre, 128, 128);
  for (int l = 0; l < 4; ++l) {
    int K = 128 * (l + 1);
    k_transpose_bf16<<<(K * 128 + T - 1) / T, T, 0, stream>>>(conv_w[l], wt_conv[l], K, 128);
  }
  k_transpose_bf16<<<(640 * 128 + T - 1) / T, T, 0, stream>>>(post_w1, wt_p1, 640, 128);
  k_transpose_bf16<<<(128 * 128 + T - 1) / T, T, 0, stream>>>(post_w2, wt_p2, 128, 128);
  k_transpose_bf16<<<(128 * 256 + T - 1) / T, T, 0, stream>>>(post_w3, wt_p3, 128, 256);
  k_transpose_bf16<<<(256 * 128 + T - 1) / T, T, 0, stream>>>(post_w4, wt_p4, 256, 128);

  // --- symmetric normalization (self-loop -> deg init 1.0) ---
  k_fill_f32<<<(NN + T - 1) / T, T, 0, stream>>>(deg, 1.0f, NN);
  k_degree<<<(NE + T - 1) / T, T, 0, stream>>>(ei, deg, NE);
  k_dinv<<<(NN + T - 1) / T, T, 0, stream>>>(deg, dinv, NN);

  dim3 g1((NN + 127) / 128, 1);
  dim3 g2((NN + 127) / 128, 2);

  // --- pre_mp: emb0 = x @ pre_w + pre_b  -> embh[:, 0:128] ---
  k_gemm_bf16_wmma<<<g1, 256, 0, stream>>>(xh, HID, wt_pre, pre_b,
                                           nullptr, 0, embh, 640, NN, HID, 0);

  // --- 4 GCN layers with skip-concat ---
  for (int l = 0; l < 4; ++l) {
    int K = 128 * (l + 1);
    // h_lin = emb[:, :K] @ conv_w[l]
    k_gemm_bf16_wmma<<<g1, 256, 0, stream>>>(embh, 640, wt_conv[l], nullptr,
                                             hlin, HID, nullptr, 0, NN, K, 0);
    // agg = dinv^2 * h_lin (self loops), then scatter neighbor messages
    k_agg_init<<<eb, T, 0, stream>>>(hlin, dinv, agg, nElem);
    k_scatter<<<NE / 8, 256, 0, stream>>>(ei, hlin, dinv, agg, NE);
    // emb[:, K:K+128] = relu(agg + b)
    k_finalize<<<eb, T, 0, stream>>>(agg, conv_b[l], embh, K, nElem);
  }

  // --- post_mp ---
  k_gemm_bf16_wmma<<<g1, 256, 0, stream>>>(embh, 640, wt_p1, post_b1,
                                           nullptr, 0, t1h, HID, NN, 640, 2); // LeakyReLU
  k_gemm_bf16_wmma<<<g1, 256, 0, stream>>>(t1h, HID, wt_p2, post_b2,
                                           nullptr, 0, t2h, HID, NN, HID, 1); // ReLU
  k_gemm_bf16_wmma<<<g2, 256, 0, stream>>>(t2h, HID, wt_p3, post_b3,
                                           nullptr, 0, t3h, 256, NN, HID, 1); // ReLU, N=256
  k_gemm_bf16_wmma<<<g1, 256, 0, stream>>>(t3h, 256, wt_p4, post_b4,
                                           (float*)d_out, HID, nullptr, 0, NN, 256, 0);
}
